// MultiHeadAttention_386547056757
// MI455X (gfx1250) — matrix operand order repaired and checked by reading, not compiled, not run on hardware
//
#include <hip/hip_runtime.h>
#include <hip/hip_bf16.h>
#include <math.h>

// MI455X / gfx1250, wave32. All matrix math via v_wmma_f32_16x16x32_f16.
// Software-pipelined: double-buffered LDS tiles in GEMMs, register-staged
// K/V tiles in attention so global loads overlap WMMA / softmax phases.

typedef __attribute__((ext_vector_type(16))) _Float16 v16h;
typedef __attribute__((ext_vector_type(8)))  _Float16 v8h;
typedef __attribute__((ext_vector_type(8)))  float    v8f;

#define SEQ  2048
#define NE   2048
#define NH   16
#define DIM  128
#define D3   384   // 3*DIM

// ---------------------------------------------------------------------------
// Fragment helpers (CDNA5 WMMA 16x16x32 f16, wave32 layouts per ISA 7.12.2)
// ---------------------------------------------------------------------------

// A fragment: 16x32 (MxK), source row-major with leading dim ld.
// lanes 0-15: row=lane,  K = {0..7,16..23}; lanes 16-31: row=lane-16, K+8.
__device__ __forceinline__ v16h load_a_frag(const _Float16* p, int ld) {
  const int lane = threadIdx.x & 31;
  const int r = lane & 15, hi = lane >> 4;
  const _Float16* row = p + (size_t)r * ld;
  v16h a;
#pragma unroll
  for (int j = 0; j < 8; ++j) {
    int k = ((j < 4) ? (2 * j) : (2 * j + 8)) + hi * 8;
    a[2 * j]     = row[k];
    a[2 * j + 1] = row[k + 1];
  }
  return a;
}

// B fragment: 32x16 (KxN), source stored TRANSPOSED as [N][K] row-major (ld).
// lanes 0-15: col=lane, K = {0..7,16..23}; lanes 16-31: col=lane-16, K+8.
__device__ __forceinline__ v16h load_bt_frag(const _Float16* p, int ld) {
  const int lane = threadIdx.x & 31;
  const int n = lane & 15, hi = lane >> 4;
  const _Float16* row = p + (size_t)n * ld + hi * 8;
  v16h b;
#pragma unroll
  for (int j = 0; j < 8; ++j) {
    int kb = (j < 4) ? (2 * j) : (2 * j + 8);
    b[2 * j]     = row[kb];
    b[2 * j + 1] = row[kb + 1];
  }
  return b;
}

__device__ __forceinline__ v8f wmma_f16(v16h a, v16h b, v8f c) {
  return __builtin_amdgcn_wmma_f32_16x16x32_f16(false, a, false, b,
                                                (short)0, c, false, false);
}

// ---------------------------------------------------------------------------
// Conversion kernels
// ---------------------------------------------------------------------------
__global__ void cvt_kernel(const float* __restrict__ src,
                           _Float16* __restrict__ dst, int n) {
  int i = blockIdx.x * 256 + threadIdx.x;
  if (i < n) dst[i] = (_Float16)src[i];
}

// src[b][rows][cols] (f32) -> dst[b][cols][rows] (f16)
__global__ void cvt_t_kernel(const float* __restrict__ src,
                             _Float16* __restrict__ dst, int rows, int cols) {
  int b = blockIdx.y;
  int i = blockIdx.x * 256 + threadIdx.x;
  int n = rows * cols;
  if (i < n) {
    int r = i / cols, c = i % cols;
    dst[(size_t)b * n + (size_t)c * rows + r] = (_Float16)src[(size_t)b * n + i];
  }
}

// ---------------------------------------------------------------------------
// Shared GEMM building blocks: 128x128 tile, 8 waves (2x4), K-step 64,
// double-buffered LDS (stride 72 halves -> 16B-aligned b128 LDS ops).
// ---------------------------------------------------------------------------
__device__ __forceinline__ void load_tile_regs(v8h* ra, v8h* rb,
    const _Float16* __restrict__ A, const _Float16* __restrict__ Bt,
    int m0, int n0, int k0, int tid) {
#pragma unroll
  for (int t = 0; t < 4; ++t) {
    int v = tid + 256 * t;              // 0..1023
    int row = v >> 3, cv = (v & 7) * 8; // 128 rows x 64 cols
    ra[t] = *(const v8h*)&A [(size_t)(m0 + row) * NE + k0 + cv];
    rb[t] = *(const v8h*)&Bt[(size_t)(n0 + row) * NE + k0 + cv];
  }
}

__device__ __forceinline__ void store_tile_lds(_Float16 (*As)[72],
    _Float16 (*Bs)[72], const v8h* ra, const v8h* rb, int tid) {
#pragma unroll
  for (int t = 0; t < 4; ++t) {
    int v = tid + 256 * t;
    int row = v >> 3, cv = (v & 7) * 8;
    *(v8h*)&As[row][cv] = ra[t];
    *(v8h*)&Bs[row][cv] = rb[t];
  }
}

__device__ __forceinline__ void compute_tile(v8f acc[4][2],
    const _Float16 (*As)[72], const _Float16 (*Bs)[72], int wm, int wn) {
#pragma unroll
  for (int kk = 0; kk < 2; ++kk) {
    v16h af[4];
#pragma unroll
    for (int mi = 0; mi < 4; ++mi)
      af[mi] = load_a_frag(&As[wm * 64 + mi * 16][kk * 32], 72);
#pragma unroll
    for (int ni = 0; ni < 2; ++ni) {
      v16h bf = load_bt_frag(&Bs[wn * 32 + ni * 16][kk * 32], 72);
#pragma unroll
      for (int mi = 0; mi < 4; ++mi)
        acc[mi][ni] = wmma_f16(af[mi], bf, acc[mi][ni]);
    }
  }
}

// ---------------------------------------------------------------------------
// QKV projection: per head, [2048x2048] @ [2048x384] + bias.
// ---------------------------------------------------------------------------
__global__ __launch_bounds__(256) void qkv_gemm_kernel(
    const _Float16* __restrict__ H, const _Float16* __restrict__ WT,
    const float* __restrict__ bias, _Float16* __restrict__ q,
    _Float16* __restrict__ kbuf, _Float16* __restrict__ vT) {
  __shared__ _Float16 As[2][128][72];
  __shared__ _Float16 Bs[2][128][72];
  const int h  = blockIdx.z;
  const int m0 = blockIdx.x * 128;
  const int n0 = blockIdx.y * 128;
  const _Float16* Bt = WT + (size_t)h * D3 * NE;
  const int tid = threadIdx.x;
  const int w = tid >> 5, lane = tid & 31;
  const int wm = w >> 2, wn = w & 3;
  const int nl = lane & 15, hi = lane >> 4;

  v8f acc[4][2] = {};
  v8h ra[4], rb[4];

  load_tile_regs(ra, rb, H, Bt, m0, n0, 0, tid);
  store_tile_lds(As[0], Bs[0], ra, rb, tid);
  const int nIter = NE / 64;  // 32 (even)
  for (int i = 0; i < nIter; i += 2) {
    // stage A: compute buf0, prefetch i+1 -> buf1
    load_tile_regs(ra, rb, H, Bt, m0, n0, (i + 1) * 64, tid);
    __syncthreads();
    compute_tile(acc, As[0], Bs[0], wm, wn);
    store_tile_lds(As[1], Bs[1], ra, rb, tid);
    // stage B: compute buf1, prefetch i+2 -> buf0
    if (i + 2 < nIter)
      load_tile_regs(ra, rb, H, Bt, m0, n0, (i + 2) * 64, tid);
    __syncthreads();
    compute_tile(acc, As[1], Bs[1], wm, wn);
    if (i + 2 < nIter)
      store_tile_lds(As[0], Bs[0], ra, rb, tid);
  }

  const float* bq = bias + h * D3;
#pragma unroll
  for (int mi = 0; mi < 4; ++mi) {
#pragma unroll
    for (int ni = 0; ni < 2; ++ni) {
      int col = n0 + wn * 32 + ni * 16 + nl;   // 0..383
      float b = bq[col];
#pragma unroll
      for (int r = 0; r < 8; ++r) {
        int row = m0 + wm * 64 + mi * 16 + hi * 8 + r;
        _Float16 hv = (_Float16)(acc[mi][ni][r] + b);
        if (col < DIM)
          q[((size_t)h * SEQ + row) * DIM + col] = hv;
        else if (col < 2 * DIM)
          kbuf[((size_t)h * SEQ + row) * DIM + (col - DIM)] = hv;
        else
          vT[((size_t)h * DIM + (col - 2 * DIM)) * SEQ + row] = hv;
      }
    }
  }
}

// ---------------------------------------------------------------------------
// Causal flash attention. One block = (head, 128-query tile). 8 waves,
// wave w owns query rows [16w,16w+16), full 128-wide N per wave.
// K/V tiles staged through registers so global latency overlaps compute.
// ---------------------------------------------------------------------------
__global__ __launch_bounds__(256) void attn_kernel(
    const _Float16* __restrict__ q, const _Float16* __restrict__ kbuf,
    const _Float16* __restrict__ vT, _Float16* __restrict__ ao) {
  __shared__ _Float16 KVs[128][136];  // K tile [t][d], then V tile [d][t]
  __shared__ float    Ss[128][132];   // scores f32
  __shared__ _Float16 Ps[128][136];   // probs f16
  __shared__ float rowm[128], rowl[128], rowa[128];

  const int h  = blockIdx.y;
  const int qt = blockIdx.x;
  const int q0 = qt * 128;
  const int tid = threadIdx.x;
  const int w = tid >> 5, lane = tid & 31;
  const int nl = lane & 15, hi = lane >> 4;
  const float sc = 0.02209708691207961f;  // 1/sqrt(N_EMBD)

  // Q fragments in registers (rows q0+16w .. +16, 4 K-chunks of 32)
  v16h qf[4];
  {
    const _Float16* qp = q + ((size_t)h * SEQ + q0 + 16 * w) * DIM;
#pragma unroll
    for (int kk = 0; kk < 4; ++kk) qf[kk] = load_a_frag(qp + kk * 32, DIM);
  }
  v8f o[8] = {};
  if (tid < 128) { rowm[tid] = -INFINITY; rowl[tid] = 0.f; }

  // prefetch K tile for kt=0 into registers
  v8h kr[8];
#pragma unroll
  for (int t = 0; t < 8; ++t) {
    int v = tid + 256 * t;               // 0..2047
    int row = v >> 4, cv = (v & 15) * 8; // 128x128
    kr[t] = *(const v8h*)&kbuf[((size_t)h * SEQ + row) * DIM + cv];
  }

  for (int kt = 0; kt <= qt; ++kt) {
    const int t0 = kt * 128;
    __syncthreads();            // prior P@V done reading KVs
    // K tile regs -> LDS: KVs[j][d]
#pragma unroll
    for (int t = 0; t < 8; ++t) {
      int v = tid + 256 * t;
      int row = v >> 4, cv = (v & 15) * 8;
      *(v8h*)&KVs[row][cv] = kr[t];
    }
    __syncthreads();
    // issue V tile global loads; they complete under the S-WMMA phase
    v8h vr[8];
#pragma unroll
    for (int t = 0; t < 8; ++t) {
      int v = tid + 256 * t;
      int row = v >> 4, cv = (v & 15) * 8;
      vr[t] = *(const v8h*)&vT[((size_t)h * DIM + row) * SEQ + t0 + cv];
    }
    // S = Q K^T   (B^T layout = k row-major [t][d])
    v8f sacc[8] = {};
#pragma unroll
    for (int kk = 0; kk < 4; ++kk) {
#pragma unroll
      for (int ni = 0; ni < 8; ++ni) {
        v16h bf = load_bt_frag(&KVs[ni * 16][kk * 32], 136);
        sacc[ni] = wmma_f16(qf[kk], bf, sacc[ni]);
      }
    }
#pragma unroll
    for (int ni = 0; ni < 8; ++ni)
#pragma unroll
      for (int r = 0; r < 8; ++r)
        Ss[16 * w + hi * 8 + r][ni * 16 + nl] = sacc[ni][r];
    __syncthreads();
    // online softmax: 2 threads per row (same-wave pair, shfl_xor combine)
    {
      const int i = tid >> 1, hf = tid & 1, gi = q0 + i;
      int jmax = gi - t0 + 1; if (jmax > 128) jmax = 128;
      const int j0 = hf * 64;
      int jend = jmax - j0;
      if (jend > 64) jend = 64;
      float mold = rowm[i];
      float lmax = -INFINITY;
      for (int j = 0; j < jend; ++j) lmax = fmaxf(lmax, Ss[i][j0 + j] * sc);
      float omax = __shfl_xor(lmax, 1, 32);
      float mnew = fmaxf(mold, fmaxf(lmax, omax));
      float al = __expf(mold - mnew);
      float s = 0.f;
      for (int j = 0; j < 64; ++j) {
        float e = (j < jend) ? __expf(Ss[i][j0 + j] * sc - mnew) : 0.f;
        Ps[i][j0 + j] = (_Float16)e;
        s += e;
      }
      float os = __shfl_xor(s, 1, 32);
      if (hf == 0) {
        rowl[i] = rowl[i] * al + (s + os);
        rowm[i] = mnew;
        rowa[i] = al;
      }
    }
    // prefetch next K tile (overlaps rescale + P@V WMMAs)
    if (kt < qt) {
#pragma unroll
      for (int t = 0; t < 8; ++t) {
        int v = tid + 256 * t;
        int row = v >> 4, cv = (v & 15) * 8;
        kr[t] = *(const v8h*)&kbuf[((size_t)h * SEQ + t0 + 128 + row) * DIM + cv];
      }
    }
    // V tile regs -> LDS (K reads finished before the post-S barrier)
#pragma unroll
    for (int t = 0; t < 8; ++t) {
      int v = tid + 256 * t;
      int row = v >> 4, cv = (v & 15) * 8;
      *(v8h*)&KVs[row][cv] = vr[t];
    }
    __syncthreads();
    // rescale running O, then O += P @ V
#pragma unroll
    for (int r = 0; r < 8; ++r) {
      float al = rowa[16 * w + hi * 8 + r];
#pragma unroll
      for (int ni = 0; ni < 8; ++ni) o[ni][r] *= al;
    }
#pragma unroll
    for (int kk = 0; kk < 4; ++kk) {
      v16h pf = load_a_frag(&Ps[16 * w][kk * 32], 136);
#pragma unroll
      for (int ni = 0; ni < 8; ++ni) {
        v16h bf = load_bt_frag(&KVs[ni * 16][kk * 32], 136);
        o[ni] = wmma_f16(pf, bf, o[ni]);
      }
    }
  }
  __syncthreads();
  // normalize and store: ao[s][h*DIM + d]  (== transpose(1,0,2).reshape)
#pragma unroll
  for (int r = 0; r < 8; ++r) {
    int lr = 16 * w + hi * 8 + r;
    float inv = 1.f / rowl[lr];
    int row = q0 + lr;
#pragma unroll
    for (int ni = 0; ni < 8; ++ni)
      ao[(size_t)row * NE + h * DIM + ni * 16 + nl] = (_Float16)(o[ni][r] * inv);
  }
}

// ---------------------------------------------------------------------------
// Output projection: [2048x2048] @ [2048x2048] + bout -> f32 out
// ---------------------------------------------------------------------------
__global__ __launch_bounds__(256) void out_gemm_kernel(
    const _Float16* __restrict__ A, const _Float16* __restrict__ Bt,
    const float* __restrict__ bias, float* __restrict__ out) {
  __shared__ _Float16 As[2][128][72];
  __shared__ _Float16 Bs[2][128][72];
  const int m0 = blockIdx.x * 128;
  const int n0 = blockIdx.y * 128;
  const int tid = threadIdx.x;
  const int w = tid >> 5, lane = tid & 31;
  const int wm = w >> 2, wn = w & 3;
  const int nl = lane & 15, hi = lane >> 4;

  v8f acc[4][2] = {};
  v8h ra[4], rb[4];

  load_tile_regs(ra, rb, A, Bt, m0, n0, 0, tid);
  store_tile_lds(As[0], Bs[0], ra, rb, tid);
  const int nIter = NE / 64;  // 32 (even)
  for (int i = 0; i < nIter; i += 2) {
    load_tile_regs(ra, rb, A, Bt, m0, n0, (i + 1) * 64, tid);
    __syncthreads();
    compute_tile(acc, As[0], Bs[0], wm, wn);
    store_tile_lds(As[1], Bs[1], ra, rb, tid);
    if (i + 2 < nIter)
      load_tile_regs(ra, rb, A, Bt, m0, n0, (i + 2) * 64, tid);
    __syncthreads();
    compute_tile(acc, As[1], Bs[1], wm, wn);
    if (i + 2 < nIter)
      store_tile_lds(As[0], Bs[0], ra, rb, tid);
  }

#pragma unroll
  for (int mi = 0; mi < 4; ++mi) {
#pragma unroll
    for (int ni = 0; ni < 2; ++ni) {
      int col = n0 + wn * 32 + ni * 16 + nl;
      float b = bias[col];
#pragma unroll
      for (int r = 0; r < 8; ++r) {
        int row = m0 + wm * 64 + mi * 16 + hi * 8 + r;
        out[(size_t)row * NE + col] = acc[mi][ni][r] + b;
      }
    }
  }
}

// ---------------------------------------------------------------------------
// Workspace layout (f16 elements):
//   Hh      @ 0          (4,194,304)   hidden f16
//   WqkvT   @ 4194304    (12,582,912)  [h][384][2048]
//   WoutT   @ 16777216   (4,194,304)   [n][k]
//   q       @ 20971520   (4,194,304)   [h][s][128]
//   k       @ 25165824   (4,194,304)   [h][t][128]
//   vT      @ 29360128   (4,194,304)   [h][128][t]
//   attnOut @ 33554432   (4,194,304)   [s][2048]
// total 37,748,736 halves = 75,497,472 bytes
// ---------------------------------------------------------------------------
extern "C" void kernel_launch(void* const* d_in, const int* in_sizes, int n_in,
                              void* d_out, int out_size, void* d_ws,
                              size_t ws_size, hipStream_t stream) {
  const float* hidden = (const float*)d_in[0];
  const float* Wqkv   = (const float*)d_in[1];
  const float* bqkv   = (const float*)d_in[2];
  const float* Wout   = (const float*)d_in[3];
  const float* bout   = (const float*)d_in[4];
  float* out = (float*)d_out;

  _Float16* ws    = (_Float16*)d_ws;
  _Float16* Hh    = ws;
  _Float16* WqkvT = ws + 4194304;
  _Float16* WoutT = ws + 16777216;
  _Float16* qb    = ws + 20971520;
  _Float16* kb    = ws + 25165824;
  _Float16* vT    = ws + 29360128;
  _Float16* ao    = ws + 33554432;

  cvt_kernel<<<(SEQ * NE) / 256, 256, 0, stream>>>(hidden, Hh, SEQ * NE);
  cvt_t_kernel<<<dim3((NE * D3 + 255) / 256, NH), 256, 0, stream>>>(
      Wqkv, WqkvT, NE, D3);
  cvt_t_kernel<<<dim3((NE * NE + 255) / 256, 1), 256, 0, stream>>>(
      Wout, WoutT, NE, NE);
  qkv_gemm_kernel<<<dim3(SEQ / 128, D3 / 128, NH), 256, 0, stream>>>(
      Hh, WqkvT, bqkv, qb, kb, vT);
  attn_kernel<<<dim3(SEQ / 128, NH), 256, 0, stream>>>(qb, kb, vT, ao);
  out_gemm_kernel<<<dim3(SEQ / 128, NE / 128), 256, 0, stream>>>(
      ao, WoutT, bout, out);
}
